// LeafBlockAttention_17609365913919
// MI455X (gfx1250) — matrix operand order repaired and checked by reading, not compiled, not run on hardware
//
#include <hip/hip_runtime.h>
#include <hip/hip_bf16.h>
#include <math.h>

typedef __attribute__((ext_vector_type(16))) __bf16 v16bf;
typedef __attribute__((ext_vector_type(8)))  float  v8f;
typedef __attribute__((ext_vector_type(2)))  float  v2f;

constexpr int Bx = 4, Nx = 32768, Cx = 128, Lx = 16, Hx = 8;
constexpr int NBx   = Nx / Lx;          // 2048
constexpr int NBLK  = Bx * NBx;         // 8192 blocks
constexpr int WQ_ELEMS = 3 * Cx * Cx;   // 49152
constexpr int WP_ELEMS = Cx * Cx;       // 16384

__device__ __forceinline__ v8f wmma_bf16(v16bf a, v16bf b, v8f c) {
  return __builtin_amdgcn_wmma_f32_16x16x32_bf16(false, a, false, b, (short)0, c, false, false);
}
__device__ __forceinline__ v8f wmma_f32(v2f a, v2f b, v8f c) {
  return __builtin_amdgcn_wmma_f32_16x16x4_f32(false, a, false, b, (short)0, c, false, false);
}

// ---- pre-pass: fp32 weights -> bf16 hi/lo pairs (lives in L2 for whole grid)
__global__ void cvt_weights(const float* __restrict__ wq, const float* __restrict__ wp,
                            __bf16* __restrict__ qh, __bf16* __restrict__ ql,
                            __bf16* __restrict__ ph, __bf16* __restrict__ pl) {
  int i = blockIdx.x * blockDim.x + threadIdx.x;
  if (i < WQ_ELEMS) {
    float v = wq[i]; __bf16 h = (__bf16)v;
    qh[i] = h; ql[i] = (__bf16)(v - (float)h);
  } else {
    int j = i - WQ_ELEMS;
    if (j < WP_ELEMS) {
      float v = wp[j]; __bf16 h = (__bf16)v;
      ph[j] = h; pl[j] = (__bf16)(v - (float)h);
    }
  }
}

// ---- one workgroup (8 waves) per (b, nb) block
__global__ __launch_bounds__(256)
void leaf_attn(const float* __restrict__ x, const int* __restrict__ amask,
               const float* __restrict__ edge,
               const float* __restrict__ bqkv, const float* __restrict__ bproj,
               const float* __restrict__ egw1, const float* __restrict__ egb1,
               const float* __restrict__ egw2, const float* __restrict__ egb2,
               const float* __restrict__ wbr,  const float* __restrict__ bbr,
               const __bf16* __restrict__ wq_h, const __bf16* __restrict__ wq_l,
               const __bf16* __restrict__ wp_h, const __bf16* __restrict__ wp_l,
               float* __restrict__ out) {
  __shared__ float  s_x[2048];           // xb fp32 (16 x 128)
  __shared__ __bf16 s_xh[2048];          // xb / x_mid hi (reused)
  __shared__ __bf16 s_xl[2048];          // xb / x_mid lo (reused)
  __shared__ float  s_qkv[16 * 384];     // qkv fp32
  __shared__ float  s_sc[8 * 256];       // scores per head
  __shared__ float  s_ew[256 * 8];       // edge weights per pair per head
  __shared__ float  s_bp[256 * 4];       // bias-physics (diag-fixed)
  __shared__ int    s_mask[256];
  __shared__ float  s_need[16];

  const int tid  = threadIdx.x;
  const int blk  = blockIdx.x;           // b*NB + nb
  // wave id as a *scalar* so tile loops are provably wave-uniform (EXEC stays all-1s)
  const int wv   = __builtin_amdgcn_readfirstlane(tid >> 5);
  const int lane = tid & 31;
  const int n16  = lane & 15;
  const int half = lane >> 4;

  const size_t xbase = (size_t)blk * 2048;   // xb is 2048 contiguous floats
  const size_t ebase = (size_t)blk * 256;

  // warm the caches: this wave's Wqkv tile rows (48 rows of 256B each, hi+lo)
  // and its Wproj tile rows, issued before any dependent work.
  {
    int r0 = wv * 48;
#pragma unroll
    for (int rr = 0; rr < 2; ++rr) {
      int r = lane + rr * 32;
      if (r < 48) {
        __builtin_prefetch(wq_h + (size_t)(r0 + r) * 128, 0, 3);
        __builtin_prefetch(wq_l + (size_t)(r0 + r) * 128, 0, 3);
      }
    }
    if (lane < 16) {
      __builtin_prefetch(wp_h + (size_t)(wv * 16 + lane) * 128, 0, 3);
      __builtin_prefetch(wp_l + (size_t)(wv * 16 + lane) * 128, 0, 3);
    }
  }

  // ---- stage x (fp32 + bf16 hi/lo), mask, bias-physics into LDS
  {
    const float4* xv = reinterpret_cast<const float4*>(x + xbase);
    for (int i = tid; i < 512; i += 256) {
      float4 f = xv[i];
      int e = i * 4;
      float vv[4] = {f.x, f.y, f.z, f.w};
#pragma unroll
      for (int j = 0; j < 4; ++j) {
        float v = vv[j];
        __bf16 h = (__bf16)v;
        s_x[e + j] = v; s_xh[e + j] = h; s_xl[e + j] = (__bf16)(v - (float)h);
      }
    }
    s_mask[tid] = amask[ebase + tid];
    int qi = tid >> 4, ki = tid & 15;
    float4 e;
    if (qi == ki) { e.x = 0.f; e.y = 0.f; e.z = 0.f; e.w = 1.f; }
    else e = reinterpret_cast<const float4*>(edge)[ebase + tid];
    s_bp[tid * 4 + 0] = e.x; s_bp[tid * 4 + 1] = e.y;
    s_bp[tid * 4 + 2] = e.z; s_bp[tid * 4 + 3] = e.w;
  }
  __syncthreads();

  if (tid < 16) {
    int rs = 0;
#pragma unroll
    for (int ki = 0; ki < 16; ++ki) rs += s_mask[tid * 16 + ki];
    s_need[tid] = (rs < 1) ? 1.0f : 0.0f;
  }
  __syncthreads();

  // ---- edge-gate MLP: one (qi,ki) pair per thread (exact gelu)
  {
    int qi = tid >> 4, ki = tid & 15;
    float b0 = s_bp[tid * 4 + 0], b1 = s_bp[tid * 4 + 1];
    float b2 = s_bp[tid * 4 + 2], b3 = s_bp[tid * 4 + 3];
    float mm = (float)s_mask[tid];
    if (qi == ki) mm = fmaxf(mm, s_need[qi]);
    float ew[8];
#pragma unroll
    for (int h = 0; h < 8; ++h) ew[h] = egb2[h];
#pragma unroll
    for (int j = 0; j < 16; ++j) {
      float a = egb1[j] + b0 * egw1[j * 4 + 0] + b1 * egw1[j * 4 + 1]
                        + b2 * egw1[j * 4 + 2] + b3 * egw1[j * 4 + 3];
      float g = 0.5f * a * (1.0f + erff(a * 0.70710678118654752f));
#pragma unroll
      for (int h = 0; h < 8; ++h) ew[h] += g * egw2[h * 16 + j];
    }
#pragma unroll
    for (int h = 0; h < 8; ++h) s_ew[tid * 8 + h] = (mm == 0.0f) ? 0.0f : ew[h];
  }

  // ---- qkv GEMM: D(16x16) tiles, bf16-split-3 WMMA, 3 tiles per wave (unrolled)
#pragma unroll
  for (int tt = 0; tt < 3; ++tt) {
    const int t = wv * 3 + tt;
    v8f acc = {};
#pragma unroll
    for (int kc = 0; kc < 4; ++kc) {
      v16bf ah, al;
      int ab = n16 * 128 + kc * 32 + half * 16;
#pragma unroll
      for (int i = 0; i < 8; ++i) {
        ah[i]     = s_xh[ab + i];      al[i]     = s_xl[ab + i];
        ah[8 + i] = s_xh[ab + 8 + i]; al[8 + i] = s_xl[ab + 8 + i];
      }
      size_t bb = (size_t)(t * 16 + n16) * 128 + kc * 32 + half * 16;
      v16bf bh = *reinterpret_cast<const v16bf*>(wq_h + bb);
      v16bf bl = *reinterpret_cast<const v16bf*>(wq_l + bb);
      acc = wmma_bf16(ah, bh, acc);
      acc = wmma_bf16(al, bh, acc);
      acc = wmma_bf16(ah, bl, acc);
    }
    float bias = bqkv[t * 16 + n16];
#pragma unroll
    for (int r = 0; r < 8; ++r)
      s_qkv[(half * 8 + r) * 384 + t * 16 + n16] = acc[r] + bias;
  }
  __syncthreads();

  // ---- attention scores: one head per wave, exact fp32 WMMA 16x16x4
  {
    const int h = wv;
    v8f sc = {};
#pragma unroll
    for (int s = 0; s < 4; ++s) {
      v2f a, b;
      int ka = h * 16 + 4 * s + half * 2;
      a[0] = s_qkv[n16 * 384 + ka];       a[1] = s_qkv[n16 * 384 + ka + 1];
      b[0] = s_qkv[n16 * 384 + 128 + ka]; b[1] = s_qkv[n16 * 384 + 128 + ka + 1];
      sc = wmma_f32(a, b, sc);
    }
#pragma unroll
    for (int r = 0; r < 8; ++r)
      s_sc[h * 256 + (half * 8 + r) * 16 + n16] = sc[r] * 0.25f;   // * D^-0.5
  }
  __syncthreads();

  // ---- softmax + edge-mix + block gate: one (head, row) per thread
  if (tid < 128) {
    const int h = tid >> 4, qi = tid & 15;
    float att[16], mx = -__builtin_inff();
#pragma unroll
    for (int ki = 0; ki < 16; ++ki) {
      int p = qi * 16 + ki;
      float mm = (float)s_mask[p];
      if (ki == qi) mm = fmaxf(mm, s_need[qi]);
      float logit = (mm == 0.0f) ? -__builtin_inff() : s_bp[p * 4 + 3];
      float v = s_sc[h * 256 + p] + logit;
      att[ki] = v; mx = fmaxf(mx, v);
    }
    float sum = 0.0f;
#pragma unroll
    for (int ki = 0; ki < 16; ++ki) { float e = expf(att[ki] - mx); att[ki] = e; sum += e; }
    float inv = 1.0f / sum;

    float xm[16];
#pragma unroll
    for (int d = 0; d < 16; ++d) xm[d] = 0.0f;
#pragma unroll
    for (int ki = 0; ki < 16; ++ki) {
      float coef = att[ki] * inv + s_ew[(qi * 16 + ki) * 8 + h];   // softmax + linear edge mix
      const float* vr = &s_qkv[ki * 384 + 256 + h * 16];
#pragma unroll
      for (int d = 0; d < 16; ++d) xm[d] += coef * vr[d];
    }
    // sigmoid block gate
    float ga = bbr[h];
    for (int c = 0; c < 128; ++c) ga += s_x[qi * 128 + c] * wbr[h * 128 + c];
    float g = 1.0f / (1.0f + expf(-ga));
    // v_blk = mean over tokens of v (bias is affine -> exact)
#pragma unroll
    for (int d = 0; d < 16; ++d) {
      float vb = 0.0f;
#pragma unroll
      for (int l = 0; l < 16; ++l) vb += s_qkv[l * 384 + 256 + h * 16 + d];
      xm[d] += g * vb * 0.0625f;
    }
    // store x_mid as bf16 hi/lo (reuse s_xh/s_xl)
#pragma unroll
    for (int d = 0; d < 16; ++d) {
      int o = qi * 128 + h * 16 + d;
      __bf16 hh = (__bf16)xm[d];
      s_xh[o] = hh; s_xl[o] = (__bf16)(xm[d] - (float)hh);
    }
  }
  __syncthreads();

  // ---- output projection: 1 tile per wave, bf16-split-3 WMMA, write global
  {
    const int t = wv;
    v8f acc = {};
#pragma unroll
    for (int kc = 0; kc < 4; ++kc) {
      v16bf ah, al;
      int ab = n16 * 128 + kc * 32 + half * 16;
#pragma unroll
      for (int i = 0; i < 8; ++i) {
        ah[i]     = s_xh[ab + i];      al[i]     = s_xl[ab + i];
        ah[8 + i] = s_xh[ab + 8 + i]; al[8 + i] = s_xl[ab + 8 + i];
      }
      size_t bb = (size_t)(t * 16 + n16) * 128 + kc * 32 + half * 16;
      v16bf bh = *reinterpret_cast<const v16bf*>(wp_h + bb);
      v16bf bl = *reinterpret_cast<const v16bf*>(wp_l + bb);
      acc = wmma_bf16(ah, bh, acc);
      acc = wmma_bf16(al, bh, acc);
      acc = wmma_bf16(ah, bl, acc);
    }
    float bias = bproj[t * 16 + n16];
#pragma unroll
    for (int r = 0; r < 8; ++r) {
      int row = half * 8 + r;
      out[xbase + (size_t)row * 128 + t * 16 + n16] = acc[r] + bias;
    }
  }
}

extern "C" void kernel_launch(void* const* d_in, const int* in_sizes, int n_in,
                              void* d_out, int out_size, void* d_ws, size_t ws_size,
                              hipStream_t stream) {
  (void)in_sizes; (void)n_in; (void)out_size; (void)ws_size;
  const float* x    = (const float*)d_in[0];
  const int*   am   = (const int*)d_in[1];
  const float* ef   = (const float*)d_in[2];
  const float* wqkv = (const float*)d_in[3];
  const float* bqkv = (const float*)d_in[4];
  const float* wprj = (const float*)d_in[5];
  const float* bprj = (const float*)d_in[6];
  const float* egw1 = (const float*)d_in[7];
  const float* egb1 = (const float*)d_in[8];
  const float* egw2 = (const float*)d_in[9];
  const float* egb2 = (const float*)d_in[10];
  const float* wbr  = (const float*)d_in[11];
  const float* bbr  = (const float*)d_in[12];
  float* out = (float*)d_out;

  __bf16* qh = (__bf16*)d_ws;
  __bf16* ql = qh + WQ_ELEMS;
  __bf16* ph = ql + WQ_ELEMS;
  __bf16* pl = ph + WP_ELEMS;

  cvt_weights<<<(WQ_ELEMS + WP_ELEMS) / 256, 256, 0, stream>>>(wqkv, wprj, qh, ql, ph, pl);
  leaf_attn<<<NBLK, 256, 0, stream>>>(x, am, ef, bqkv, bprj, egw1, egb1, egw2, egb2,
                                      wbr, bbr, qh, ql, ph, pl, out);
}
